// RotaryCausalSelfAttention_26980984554237
// MI455X (gfx1250) — matrix operand order repaired and checked by reading, not compiled, not run on hardware
//
#include <hip/hip_runtime.h>
#include <hip/hip_bf16.h>

// ---------------------------------------------------------------------------
// RotaryCausalSelfAttention for MI455X (gfx1250, wave32, WMMA bf16,
// async global->LDS tile staging, double-buffered)
// B=2, T=2048, D=2048, H=16, DK=128
// ---------------------------------------------------------------------------

typedef __attribute__((ext_vector_type(16))) __bf16 v16bf;
typedef __attribute__((ext_vector_type(8)))  __bf16 v8bf;
typedef __attribute__((ext_vector_type(8)))  float  v8f;

#define BATCH  2
#define SEQ    2048
#define DMODEL 2048
#define NHEADS 16
#define DK     128
#define KDIM   2048   // inner dim of both projections

// Load a 16-element bf16 fragment as two contiguous 16-byte chunks.
__device__ __forceinline__ v16bf load_frag2(const __bf16* p0, const __bf16* p1) {
    union { v16bf v; v8bf h[2]; } u;
    u.h[0] = *(const v8bf*)p0;
    u.h[1] = *(const v8bf*)p1;
    return u.v;
}

__device__ __forceinline__ v8f bwmma(v16bf a, v16bf b, v8f c) {
    return __builtin_amdgcn_wmma_f32_16x16x32_bf16(
        false, a, false, b, (short)0, c, false, false);
}

// Async DMA of 16 bytes/lane from global memory into LDS (ASYNCcnt-tracked).
// lds_off = byte offset within the workgroup LDS allocation (single __shared__
// array per kernel -> allocation starts at offset 0).
__device__ __forceinline__ void async_b128(unsigned lds_off, const void* gptr) {
    asm volatile("global_load_async_to_lds_b128 %0, %1, off"
                 :: "v"(lds_off), "v"(gptr) : "memory");
}
__device__ __forceinline__ void wait_async0() {
    asm volatile("s_wait_asynccnt 0x0" ::: "memory");
}

// ---------------------------------------------------------------------------
// 1) fp32 -> bf16 conversion
// ---------------------------------------------------------------------------
__global__ __launch_bounds__(256) void cvt_bf16_kernel(const float* __restrict__ in,
                                                       __bf16* __restrict__ out, int n) {
    int i = blockIdx.x * 256 + threadIdx.x;
    if (i < n) out[i] = (__bf16)in[i];
}

// ---------------------------------------------------------------------------
// 2) NT GEMM: C[M][N] f32 = A[M][2048] bf16 x (B[N][2048] bf16)^T
// Block: 256 threads = 8 waves; block tile 64(M) x 128(N); K-step 32.
// Double-buffered async staging: A tile 64x32 (4KB), B tile 128x32 (8KB).
// Wave tile 32x32 = 2x2 WMMA accumulators.
// ---------------------------------------------------------------------------
template <int LDC>
__global__ __launch_bounds__(256) void gemm_bf16_nt_kernel(
    const __bf16* __restrict__ A, const __bf16* __restrict__ B, float* __restrict__ C) {
    __shared__ __align__(16) __bf16 smem[2 * 64 * 32 + 2 * 128 * 32]; // 24KB
    const int AOFF = 0;            // [s][64][32]
    const int BOFF = 2 * 64 * 32;  // [s][128][32]

    const int tid  = threadIdx.x;
    const int lane = tid & 31, wave = tid >> 5;
    const int nn = lane & 15, hi = lane >> 4;
    const int wm = wave & 1, wn = wave >> 1;            // 2 x 4 wave grid
    const int m_blk = blockIdx.y * 64;
    const int n_blk = blockIdx.x * 128;

    auto issue_stage = [&](int s, int k0) {
        {   // A tile: 64 rows x 4 x (8 bf16) = 256 chunks, one per thread
            int row = tid >> 2, ch = tid & 3;
            const __bf16* g = A + (size_t)(m_blk + row) * KDIM + k0 + ch * 8;
            async_b128((unsigned)((AOFF + s * 64 * 32 + row * 32 + ch * 8) * 2), g);
        }
#pragma unroll
        for (int rr = 0; rr < 2; ++rr) {  // B tile: 128 rows x 4 chunks = 512
            int id = tid + rr * 256;
            int row = id >> 2, ch = id & 3;
            const __bf16* g = B + (size_t)(n_blk + row) * KDIM + k0 + ch * 8;
            async_b128((unsigned)((BOFF + s * 128 * 32 + row * 32 + ch * 8) * 2), g);
        }
    };

    v8f acc[2][2] = {};
    issue_stage(0, 0);
    int s = 0;
    for (int k0 = 0; k0 < KDIM; k0 += 32) {
        wait_async0();           // this wave's stage-s DMA complete
        __syncthreads();         // everyone's stage-s DMA complete
        if (k0 + 32 < KDIM) issue_stage(s ^ 1, k0 + 32);  // overlap next DMA

        v16bf af[2], bfr[2];
#pragma unroll
        for (int i = 0; i < 2; ++i) {
            const __bf16* p = &smem[AOFF + s * 64 * 32 + (wm * 32 + i * 16 + nn) * 32 + hi * 8];
            af[i] = load_frag2(p, p + 16);     // A-layout: K = hi*8+e, 16+hi*8+e
        }
#pragma unroll
        for (int j = 0; j < 2; ++j) {
            const __bf16* p = &smem[BOFF + s * 128 * 32 + (wn * 32 + j * 16 + nn) * 32 + hi * 8];
            bfr[j] = load_frag2(p, p + 16);    // B-layout: K = hi*8+e, 16+hi*8+e
        }
#pragma unroll
        for (int i = 0; i < 2; ++i)
#pragma unroll
            for (int j = 0; j < 2; ++j)
                acc[i][j] = bwmma(af[i], bfr[j], acc[i][j]);

        __syncthreads();         // all reads of buffer s done before reuse
        s ^= 1;
    }

    // C/D layout: VGPR r -> row r + 8*hi, col nn. LDC is constexpr so the
    // r*LDC terms fold into immediate store offsets.
#pragma unroll
    for (int i = 0; i < 2; ++i)
#pragma unroll
        for (int j = 0; j < 2; ++j) {
            float* cp = C + (size_t)(m_blk + wm * 32 + i * 16 + 8 * hi) * LDC
                          + (n_blk + wn * 32 + j * 16 + nn);
#pragma unroll
            for (int r = 0; r < 8; ++r)
                cp[(size_t)r * LDC] = acc[i][j][r];
        }
}

// ---------------------------------------------------------------------------
// 3) RoPE + head split:
//    qkv f32 [B*T][3*D] -> q_bf [bh][t][128], k_bf [bh][t][128],
//                          vT_bf [bh][128][t]
// ---------------------------------------------------------------------------
__global__ __launch_bounds__(256) void rope_split_kernel(
    const float* __restrict__ qkv, const int* __restrict__ pos,
    __bf16* __restrict__ qo, __bf16* __restrict__ ko, __bf16* __restrict__ vto) {
    int idx = blockIdx.x * 256 + threadIdx.x;
    int i = idx & 63;
    int h = (idx >> 6) & 15;
    int t = (idx >> 10) & 2047;
    int b = idx >> 21;
    if (b >= BATCH) return;

    size_t row = (size_t)(b * SEQ + t) * (3 * DMODEL);
    int c0 = h * DK + 2 * i;
    float qe = qkv[row + c0],              qd = qkv[row + c0 + 1];
    float ke = qkv[row + DMODEL + c0],     kd = qkv[row + DMODEL + c0 + 1];
    float ve = qkv[row + 2 * DMODEL + c0], vd = qkv[row + 2 * DMODEL + c0 + 1];

    float p   = (float)pos[t];
    float inv = __powf(10000.0f, -(float)i * (1.0f / 64.0f));
    float ang = p * inv;
    float sn = __sinf(ang), cs = __cosf(ang);

    int bh = b * NHEADS + h;
    size_t qrow = ((size_t)bh * SEQ + t) * DK + 2 * i;
    qo[qrow]     = (__bf16)(qe * cs - qd * sn);
    qo[qrow + 1] = (__bf16)(qe * sn + qd * cs);
    ko[qrow]     = (__bf16)(ke * cs - kd * sn);
    ko[qrow + 1] = (__bf16)(ke * sn + kd * cs);

    size_t vrow = ((size_t)bh * DK + 2 * i) * SEQ + t;
    vto[vrow]       = (__bf16)ve;
    vto[vrow + SEQ] = (__bf16)vd;
}

// ---------------------------------------------------------------------------
// 4) Flash-style causal attention.
// Block = 128 threads (4 waves) = 64 queries of one (b,h); wave w owns 16.
// Per 32-key block j (uniform across waves; fully-masked blocks are exact
// no-ops since key 0 is always valid so m-state is finite after j=0):
//   - K tile 32x128 and V^T tile 128x32 async-DMA'd to LDS (double buffered)
//   - S(16x32) = q @ k^T          : 8 wmma / wave
//   - online softmax (shfl_xor reductions + v_exp_f32)
//   - P: C-layout -> A-layout through per-wave LDS tile
//   - O(16x128) += P @ V          : 8 wmma / wave
// ---------------------------------------------------------------------------
__global__ __launch_bounds__(128) void attn_kernel(
    const __bf16* __restrict__ qb, const __bf16* __restrict__ kb,
    const __bf16* __restrict__ vtb, __bf16* __restrict__ ob) {
    const float scale = 0.08838834764831845f; // 1/sqrt(128)
    __shared__ __align__(16) __bf16 smem[2 * 32 * 128 + 2 * 128 * 32 + 4 * 16 * 32];
    const int KOFF = 0;                  // [s][key(32)][d(128)]
    const int VOFF = 2 * 32 * 128;       // [s][d(128)][key(32)]
    const int POFF = VOFF + 2 * 128 * 32;// [wave][m(16)][kk(32)]

    const int tid = threadIdx.x;
    const int lane = tid & 31, w = tid >> 5;
    const int nn = lane & 15, hi = lane >> 4;
    const int blk = blockIdx.x;
    const int qt = (blk & 31) * 64;      // 64-query tile base
    const int bh = blk >> 5;             // 0..31
    const int qi0 = qt + w * 16;         // this wave's query base

    const __bf16* qbase = qb  + (size_t)bh * SEQ * DK;
    const __bf16* kbase = kb  + (size_t)bh * SEQ * DK;
    const __bf16* vbase = vtb + (size_t)bh * DK * SEQ;

    auto issue_kv = [&](int s, int kbl) {
#pragma unroll
        for (int c = 0; c < 4; ++c) {    // K tile: 32 rows x 16 chunks = 512
            int id = c * 128 + tid;
            int row = id >> 4, ch = id & 15;
            const __bf16* g = kbase + (size_t)(kbl + row) * DK + ch * 8;
            async_b128((unsigned)((KOFF + s * 32 * 128 + row * 128 + ch * 8) * 2), g);
        }
#pragma unroll
        for (int c = 0; c < 4; ++c) {    // V tile: 128 rows x 4 chunks = 512
            int id = c * 128 + tid;
            int row = id >> 2, ch = id & 3;
            const __bf16* g = vbase + (size_t)row * SEQ + kbl + ch * 8;
            async_b128((unsigned)((VOFF + s * 128 * 32 + row * 32 + ch * 8) * 2), g);
        }
    };

    // q fragments (kept in registers for the whole kernel)
    v16bf qf[4];
    {
        const __bf16* qr = qbase + (size_t)(qi0 + nn) * DK;
#pragma unroll
        for (int c = 0; c < 4; ++c) {
            const __bf16* p = qr + c * 32 + hi * 8;
            qf[c] = load_frag2(p, p + 16);
        }
    }

    v8f Oacc[8] = {};
    float mstate[8], lstate[8];
#pragma unroll
    for (int r = 0; r < 8; ++r) { mstate[r] = -1e30f; lstate[r] = 0.0f; }

    const int jlast = (qt + 63) >> 5;    // uniform across the block
    issue_kv(0, 0);
    int s = 0;
    for (int j = 0; j <= jlast; ++j) {
        const int kbl = j * 32;
        wait_async0();
        __syncthreads();                             // K/V stage s ready
        if (j < jlast) issue_kv(s ^ 1, kbl + 32);    // overlap next DMA

        // ---- S = q @ k^T ----
        v8f S0 = {}, S1 = {};
#pragma unroll
        for (int c = 0; c < 4; ++c) {
            const __bf16* p0 = &smem[KOFF + s * 32 * 128 + nn * 128 + c * 32 + hi * 8];
            const __bf16* p1 = &smem[KOFF + s * 32 * 128 + (16 + nn) * 128 + c * 32 + hi * 8];
            v16bf b0 = load_frag2(p0, p0 + 16);
            v16bf b1 = load_frag2(p1, p1 + 16);
            S0 = bwmma(qf[c], b0, S0);
            S1 = bwmma(qf[c], b1, S1);
        }

        // ---- online softmax + stage P into per-wave LDS tile ----
#pragma unroll
        for (int r = 0; r < 8; ++r) {
            int qi = qi0 + r + 8 * hi;
            float s0 = (kbl + nn      <= qi) ? S0[r] * scale : -1e30f;
            float s1 = (kbl + 16 + nn <= qi) ? S1[r] * scale : -1e30f;
            float mx = fmaxf(s0, s1);
            mx = fmaxf(mx, __shfl_xor(mx, 1));
            mx = fmaxf(mx, __shfl_xor(mx, 2));
            mx = fmaxf(mx, __shfl_xor(mx, 4));
            mx = fmaxf(mx, __shfl_xor(mx, 8));
            float mnew = fmaxf(mstate[r], mx);
            float p0 = __expf(s0 - mnew);
            float p1 = __expf(s1 - mnew);
            float rs = p0 + p1;
            rs += __shfl_xor(rs, 1);
            rs += __shfl_xor(rs, 2);
            rs += __shfl_xor(rs, 4);
            rs += __shfl_xor(rs, 8);
            float corr = __expf(mstate[r] - mnew);
            lstate[r] = lstate[r] * corr + rs;
            mstate[r] = mnew;
#pragma unroll
            for (int f = 0; f < 8; ++f) Oacc[f][r] *= corr;
            int M = r + 8 * hi;
            smem[POFF + w * 512 + M * 32 + nn]      = (__bf16)p0;
            smem[POFF + w * 512 + M * 32 + 16 + nn] = (__bf16)p1;
        }
        __syncthreads();

        // P in WMMA A-layout (m = nn, K striped by hi)
        v16bf pf;
        {
            const __bf16* lp = &smem[POFF + w * 512 + nn * 32 + hi * 8];
            pf = load_frag2(lp, lp + 16);
        }

        // ---- O += P @ V ----
#pragma unroll
        for (int f = 0; f < 8; ++f) {
            const __bf16* vp = &smem[VOFF + s * 128 * 32 + (f * 16 + nn) * 32 + hi * 8];
            v16bf vf = load_frag2(vp, vp + 16);
            Oacc[f] = bwmma(pf, vf, Oacc[f]);
        }
        __syncthreads();                  // all reads of buffer s / P done
        s ^= 1;
    }

    // ---- epilogue: O /= l, write bf16 [B*T][D] ----
    int b = bh >> 4, h = bh & 15;
#pragma unroll
    for (int r = 0; r < 8; ++r) {
        int t = qi0 + r + 8 * hi;
        float inv_l = 1.0f / lstate[r];
        size_t orow = ((size_t)(b * SEQ + t)) * DMODEL + h * DK;
#pragma unroll
        for (int f = 0; f < 8; ++f)
            ob[orow + f * 16 + nn] = (__bf16)(Oacc[f][r] * inv_l);
    }
}

// ---------------------------------------------------------------------------
// Host launcher
// ---------------------------------------------------------------------------
extern "C" void kernel_launch(void* const* d_in, const int* in_sizes, int n_in,
                              void* d_out, int out_size, void* d_ws, size_t ws_size,
                              hipStream_t stream) {
    (void)in_sizes; (void)n_in; (void)out_size; (void)ws_size;
    const float* x    = (const float*)d_in[0];   // [B,T,D]
    const int*   pos  = (const int*)d_in[1];     // [T]
    const float* Wqkv = (const float*)d_in[2];   // [3D,D]
    const float* Wo   = (const float*)d_in[3];   // [D,D]
    float* out = (float*)d_out;                  // [B,T,D]

    const int MT = BATCH * SEQ;                  // 4096 token rows
    const int N3 = 3 * DMODEL;                   // 6144

    char* ws = (char*)d_ws;
    size_t off = 0;
    auto carve = [&](size_t bytes) { void* p = ws + off; off = (off + bytes + 255) & ~(size_t)255; return p; };
    __bf16* xb     = (__bf16*)carve((size_t)MT * DMODEL * 2);
    __bf16* wqkvb  = (__bf16*)carve((size_t)N3 * DMODEL * 2);
    __bf16* wob    = (__bf16*)carve((size_t)DMODEL * DMODEL * 2);
    float*  qkvf   = (float*) carve((size_t)MT * N3 * 4);
    __bf16* qbf    = (__bf16*)carve((size_t)BATCH * NHEADS * SEQ * DK * 2);
    __bf16* kbf    = (__bf16*)carve((size_t)BATCH * NHEADS * SEQ * DK * 2);
    __bf16* vtbf   = (__bf16*)carve((size_t)BATCH * NHEADS * DK * SEQ * 2);
    __bf16* attnb  = (__bf16*)carve((size_t)MT * DMODEL * 2);

    {
        int nx = MT * DMODEL;
        int nq = N3 * DMODEL;
        int nw = DMODEL * DMODEL;
        cvt_bf16_kernel<<<(nx + 255) / 256, 256, 0, stream>>>(x, xb, nx);
        cvt_bf16_kernel<<<(nq + 255) / 256, 256, 0, stream>>>(Wqkv, wqkvb, nq);
        cvt_bf16_kernel<<<(nw + 255) / 256, 256, 0, stream>>>(Wo, wob, nw);
    }

    // qkv = x @ Wqkv^T   (M=4096, N=6144, K=2048)
    gemm_bf16_nt_kernel<6144><<<dim3(N3 / 128, MT / 64), 256, 0, stream>>>(xb, wqkvb, qkvf);

    // RoPE + head split + v transpose
    {
        int total = BATCH * SEQ * NHEADS * (DK / 2);
        rope_split_kernel<<<total / 256, 256, 0, stream>>>(qkvf, pos, qbf, kbf, vtbf);
    }

    // causal flash attention -> bf16 [B*T][D]
    attn_kernel<<<BATCH * NHEADS * (SEQ / 64), 128, 0, stream>>>(qbf, kbf, vtbf, attnb);

    // out = attn @ Wo^T   (M=4096, N=2048, K=2048)
    gemm_bf16_nt_kernel<2048><<<dim3(DMODEL / 128, MT / 64), 256, 0, stream>>>(attnb, wob, out);
}
